// NormConvTranspose2d_n_67396626808868
// MI455X (gfx1250) — matrix operand order repaired and checked by reading, not compiled, not run on hardware
//
#include <hip/hip_runtime.h>

// ---------------------------------------------------------------------------
// NormConvTranspose2d, B=4 C=64 O=64 H=W=48 K=3, stride2 pad1 outpad1 dil1.
// Output [4,64,96,96]. Per output-parity-phase the transposed conv reduces to
// a GEMM (K = C*ntaps, norm folded into bf16 weights); border (h'=95/w'=95)
// pixels recomputed exactly by a scalar fixup kernel.
// GEMM uses v_wmma_f32_16x16x32_bf16; LDS staging uses CDNA5 async
// global->LDS copies (ASYNCcnt) when the builtin is available.
// ---------------------------------------------------------------------------

typedef __attribute__((ext_vector_type(16))) __bf16       v16bf;
typedef __attribute__((ext_vector_type(8)))  float        v8f;
typedef __attribute__((ext_vector_type(4)))  unsigned int u32x4;

#if defined(__gfx1250__) && __has_builtin(__builtin_amdgcn_global_load_async_to_lds_b128)
#define HAVE_ASYNC_LDS 1
// Builtin parameter type (from clang diagnostic):
//   '__attribute__((__vector_size__(4 * sizeof(int)))) int __device__ *'
typedef int async_v4i __attribute__((vector_size(16)));
typedef __attribute__((address_space(1))) async_v4i* gas_ptr;   // global src
typedef __attribute__((address_space(3))) async_v4i* las_ptr;   // LDS dst
#endif

#define EPS_F 1e-10f

__constant__ int c_ntaps[4] = {1, 2, 2, 4};
__constant__ int c_kdim [4] = {64, 128, 128, 256};
__constant__ int c_woff [4] = {0, 4096, 12288, 20480};   // bf16-element offsets
__constant__ int c_tap  [4][4][2] = {
    { {1,1}, {0,0}, {0,0}, {0,0} },   // (even h', even w')
    { {1,0}, {1,2}, {0,0}, {0,0} },   // (even h', odd  w')
    { {0,1}, {2,1}, {0,0}, {0,0} },   // (odd  h', even w')
    { {0,0}, {0,2}, {2,0}, {2,2} }    // (odd  h', odd  w')
};

__device__ __forceinline__ unsigned short f2bf(float f) {
    unsigned int u = __float_as_uint(f);
    u = (u + 0x7FFFu + ((u >> 16) & 1u)) >> 16;   // round-to-nearest-even
    return (unsigned short)u;
}

// ---- kernel 1: x [B,C,H,W] f32 -> xb [B,H,W,C] bf16 (channel-last) --------
__global__ void prep_x_kernel(const float* __restrict__ x,
                              unsigned short* __restrict__ xb) {
    int idx = blockIdx.x * 256 + threadIdx.x;          // 589824 total, exact
    int w = idx % 48;
    int t = idx / 48;
    int h = t % 48;  t /= 48;
    int c = t % 64;
    int b = t / 64;
    xb[((b * 48 + h) * 48 + w) * 64 + c] = f2bf(x[idx]);
}

// ---- kernel 2: effective weights WeffT[ph][o][t*64+c] (norm folded) -------
__global__ void prep_w_kernel(const float* __restrict__ weight,
                              const float* __restrict__ kern,
                              unsigned short* __restrict__ wt) {
    int ph  = blockIdx.y;
    int Kd  = c_kdim[ph];
    int cnt = 64 * Kd;
    int idx = blockIdx.x * 256 + threadIdx.x;
    if (idx >= cnt) return;
    int o = idx / Kd;
    int r = idx - o * Kd;
    int t = r >> 6;
    int c = r & 63;
    int nt = c_ntaps[ph];
    const float* kb = kern + (o * 64 + c) * 9;
    float norm = EPS_F;
    for (int i = 0; i < nt; ++i)
        norm += kb[c_tap[ph][i][0] * 3 + c_tap[ph][i][1]];
    float val = weight[o * 64 + c] *
                kb[c_tap[ph][t][0] * 3 + c_tap[ph][t][1]] / norm;
    wt[c_woff[ph] + o * Kd + r] = f2bf(val);
}

// ---- kernel 3: phase GEMM via v_wmma_f32_16x16x32_bf16 --------------------
// grid (576, 4): x = 16-pixel M-tile over 9216 pixels/phase, y = phase.
// block 128 = 4 waves; wave w computes the 16x16 D tile for o in [16w,16w+16).
__global__ __launch_bounds__(128)
void gemm_kernel(const unsigned short* __restrict__ xb,
                 const unsigned short* __restrict__ wt,
                 const float* __restrict__ bias,
                 float* __restrict__ out) {
    const int ph  = blockIdx.y;
    const int Kd  = c_kdim[ph];
    const int nt  = c_ntaps[ph];
    const int phH = ph >> 1, phW = ph & 1;
    const int tid = threadIdx.x;
    const int tilebase = blockIdx.x * 16;

    __shared__ alignas(16) unsigned short Blds[64 * 256];  // WeffT[o][k], 32 KB max
    __shared__ alignas(16) unsigned short Alds[16 * 256];  // A[m][k],     8 KB max

    // Whole-phase WeffT -> LDS (layout identical; async 16B/lane copies).
    {
        const u32x4* src = (const u32x4*)(wt + c_woff[ph]);
        u32x4*       dst = (u32x4*)Blds;
        int chunks = (64 * Kd) >> 3;
        for (int i = tid; i < chunks; i += 128) {
#if HAVE_ASYNC_LDS
            __builtin_amdgcn_global_load_async_to_lds_b128(
                (gas_ptr)(src + i), (las_ptr)(dst + i), 0, 0);
#else
            dst[i] = src[i];
#endif
        }
    }
    // A tile: 16 pixels x (nt taps * 64 channels), gathered per tap.
    {
        int chunks = 16 * nt * 8;                  // 8-bf16 (16 B) chunks
        for (int i = tid; i < chunks; i += 128) {
            int m  = i / (nt * 8);
            int r  = i - m * (nt * 8);
            int t  = r >> 3;
            int c0 = (r & 7) << 3;
            int gp  = tilebase + m;
            int b   = gp / 2304;
            int rem = gp - b * 2304;
            int hh  = rem / 48;
            int ww  = rem - hh * 48;
            int hp  = 2 * hh + phH;
            int wp  = 2 * ww + phW;
            int ih  = (hp + 1 - c_tap[ph][t][0]) >> 1;   // exact by phase parity
            int iw  = (wp + 1 - c_tap[ph][t][1]) >> 1;
            unsigned short* adst = Alds + m * Kd + (t << 6) + c0;
            if (ih < 48 && iw < 48) {
                const unsigned short* asrc =
                    xb + (((b * 48 + ih) * 48 + iw) << 6) + c0;
#if HAVE_ASYNC_LDS
                __builtin_amdgcn_global_load_async_to_lds_b128(
                    (gas_ptr)asrc, (las_ptr)adst, 0, 0);
#else
                *(u32x4*)adst = *(const u32x4*)asrc;
#endif
            } else {                                     // h'=95/w'=95 tap drop
                u32x4 z = {0u, 0u, 0u, 0u};
                *(u32x4*)adst = z;
            }
        }
    }
#if HAVE_ASYNC_LDS
    asm volatile("s_wait_asynccnt 0" ::: "memory");
#endif
    __syncthreads();

    const int wave = tid >> 5;
    const int lane = tid & 31;
    const int row  = lane & 15;
    const int g    = lane >> 4;

    v8f acc = {0.f, 0.f, 0.f, 0.f, 0.f, 0.f, 0.f, 0.f};
    const u32x4* Arow = (const u32x4*)(Alds + row * Kd);
    const u32x4* Brow = (const u32x4*)(Blds + (wave * 16 + row) * Kd);

    for (int k0 = 0; k0 < Kd; k0 += 32) {
        union { u32x4 u[2]; v16bf v; } fa, fb;
        // A 16x32 bf16 frag: group0 lanes hold K{0..7,16..23}, group1 K{8..15,24..31}
        int ab = (k0 >> 3) + g;
        fa.u[0] = Arow[ab];
        fa.u[1] = Arow[ab + 2];
        // B 32x16 bf16 frag: lane column n=row, group0 holds K{0..7,16..23}, group1 K{8..15,24..31}
        int bb = (k0 >> 3) + g;
        fb.u[0] = Brow[bb];
        fb.u[1] = Brow[bb + 2];
        acc = __builtin_amdgcn_wmma_f32_16x16x32_bf16(
            false, fa.v, false, fb.v, (short)0, acc, false, false);
    }

    // D layout: lane n = row, rows M = 8*g + j in VGPR j.
    int o = wave * 16 + row;
    float bv = bias[o];
    for (int j = 0; j < 8; ++j) {
        int m   = g * 8 + j;
        int gp  = tilebase + m;
        int b   = gp / 2304;
        int rem = gp - b * 2304;
        int hh  = rem / 48;
        int ww  = rem - hh * 48;
        int hp  = 2 * hh + phH;
        int wp  = 2 * ww + phW;
        out[((b * 64 + o) * 96 + hp) * 96 + wp] = acc[j] + bv;
    }
}

// ---- kernel 4: exact fp32 recompute of the h'=95 / w'=95 border -----------
__global__ void fixup_kernel(const float* __restrict__ x,
                             const float* __restrict__ weight,
                             const float* __restrict__ kern,
                             const float* __restrict__ bias,
                             float* __restrict__ out) {
    int pid = blockIdx.x;          // 0..190
    int b   = blockIdx.y;
    int hp, wp;
    if (pid < 96) { hp = 95; wp = pid; }
    else          { wp = 95; hp = pid - 96; }
    int o = threadIdx.x;

    float acc = 0.f;
    for (int c = 0; c < 64; ++c) {
        const float* kb = kern + (o * 64 + c) * 9;
        float num = 0.f, den = EPS_F;
        for (int th = 0; th < 3; ++th) {
            int dh = hp + 1 - th;
            if (dh & 1) continue;
            int ih = dh >> 1;
            if (ih >= 48) continue;
            for (int tw = 0; tw < 3; ++tw) {
                int dw = wp + 1 - tw;
                if (dw & 1) continue;
                int iw = dw >> 1;
                if (iw >= 48) continue;
                float kv = kb[th * 3 + tw];
                den += kv;
                num += kv * x[((b * 64 + c) * 48 + ih) * 48 + iw];
            }
        }
        acc += weight[o * 64 + c] * num / den;
    }
    out[((b * 64 + o) * 96 + hp) * 96 + wp] = acc + bias[o];
}

extern "C" void kernel_launch(void* const* d_in, const int* in_sizes, int n_in,
                              void* d_out, int out_size, void* d_ws, size_t ws_size,
                              hipStream_t stream) {
    const float* x      = (const float*)d_in[0];   // [4,64,48,48]
    const float* weight = (const float*)d_in[1];   // [64,64]
    const float* kern   = (const float*)d_in[2];   // [64,64,3,3]
    const float* bias   = (const float*)d_in[3];   // [64]
    float* out = (float*)d_out;                    // [4,64,96,96]

    unsigned short* xb = (unsigned short*)d_ws;          // 589824 bf16 (1.18 MB)
    unsigned short* wt = xb + 589824;                    // 36864  bf16 (72 KB)

    prep_x_kernel<<<2304, 256, 0, stream>>>(x, xb);
    prep_w_kernel<<<dim3(64, 4), 256, 0, stream>>>(weight, kern, wt);
    gemm_kernel<<<dim3(576, 4), 128, 0, stream>>>(xb, wt, bias, out);
    fixup_kernel<<<dim3(191, 4), 64, 0, stream>>>(x, weight, kern, bias, out);
}
